// class_self_MultiHeadAttention_54065048322415
// MI455X (gfx1250) — matrix operand order repaired and checked by reading, not compiled, not run on hardware
//
#include <hip/hip_runtime.h>
#include <hip/hip_bf16.h>

// ---------------- problem constants ----------------
#define N_TOK  4096
#define DMODEL 256
#define NEXP   9
#define NHEAD  4
#define DPH    64          // dim per head
#define ATT_SCALE 0.25f    // (dph // num_heads)^-0.5 = 16^-0.5
#define LN_EPS 1e-5f

typedef __attribute__((ext_vector_type(16))) _Float16 v16h;
typedef __attribute__((ext_vector_type(8)))  float    v8f;

union FragH { v16h v; unsigned int u[8]; _Float16 h[16]; };

// A-matrix (16x32, f16) per-lane K index: VGPR j, half = lane/16
__device__ __forceinline__ int kA(int j, int half) {
  return ((j >> 2) << 4) + (half << 3) + ((j & 3) << 1);
}
// B-matrix (32x16, f16) per-lane K index
__device__ __forceinline__ int kB(int j, int half) {
  return ((j >> 2) << 4) + (half << 3) + ((j & 3) << 1);
}

__device__ __forceinline__ float redmax16(float v) {
  v = fmaxf(v, __shfl_xor(v, 1, 32));
  v = fmaxf(v, __shfl_xor(v, 2, 32));
  v = fmaxf(v, __shfl_xor(v, 4, 32));
  v = fmaxf(v, __shfl_xor(v, 8, 32));
  return v;
}
__device__ __forceinline__ float redsum16(float v) {
  v += __shfl_xor(v, 1, 32);
  v += __shfl_xor(v, 2, 32);
  v += __shfl_xor(v, 4, 32);
  v += __shfl_xor(v, 8, 32);
  return v;
}
__device__ __forceinline__ float redsum32(float v) {
  v += __shfl_xor(v, 1, 32);
  v += __shfl_xor(v, 2, 32);
  v += __shfl_xor(v, 4, 32);
  v += __shfl_xor(v, 8, 32);
  v += __shfl_xor(v, 16, 32);
  return v;
}

#define WMMA_F16(A, B, C) \
  __builtin_amdgcn_wmma_f32_16x16x32_f16(false, (A), false, (B), (short)0, (C), false, false)

// ---------------- kernel 1: convert inputs to f16 (weights transposed) ----
__global__ void prep_kernel(const float* __restrict__ x,
                            const float* __restrict__ Wq,
                            const float* __restrict__ Wk,
                            const float* __restrict__ Wv,
                            const float* __restrict__ Wf,
                            _Float16* __restrict__ xh,
                            _Float16* __restrict__ wqT,
                            _Float16* __restrict__ wkT,
                            _Float16* __restrict__ wvT,
                            _Float16* __restrict__ wfT) {
  int idx = blockIdx.x * blockDim.x + threadIdx.x;
  if (idx < N_TOK * DMODEL) xh[idx] = (_Float16)x[idx];
  if (idx < NEXP * DMODEL * DMODEL) {
    int e = idx / (DMODEL * DMODEL);
    int r = idx - e * DMODEL * DMODEL;
    int k = r / DMODEL;           // input dim
    int n = r - k * DMODEL;       // output dim
    int t = e * DMODEL * DMODEL + n * DMODEL + k;   // transposed: [e][n][k]
    wqT[t] = (_Float16)Wq[idx];
    wkT[t] = (_Float16)Wk[idx];
    wvT[t] = (_Float16)Wv[idx];
  }
  if (idx < DMODEL * DMODEL) {
    int k = idx / DMODEL, n = idx - k * DMODEL;
    wfT[n * DMODEL + k] = (_Float16)Wf[idx];
  }
}

// ---------------- kernel 2: deterministic counting sort by label ----------
__global__ void sort_kernel(const int* __restrict__ label,
                            int* __restrict__ perm,
                            int* __restrict__ offs) {
  if (threadIdx.x != 0 || blockIdx.x != 0) return;
  int cnt[NEXP];
  for (int e = 0; e < NEXP; ++e) cnt[e] = 0;
  for (int n = 0; n < N_TOK; ++n) {
    int e = label[n];
    e = (e < 0) ? 0 : ((e >= NEXP) ? (NEXP - 1) : e);
    cnt[e]++;
  }
  int run = 0;
  for (int e = 0; e < NEXP; ++e) { offs[e] = run; run += cnt[e]; }
  offs[NEXP] = run;
  int cur[NEXP];
  for (int e = 0; e < NEXP; ++e) cur[e] = offs[e];
  for (int n = 0; n < N_TOK; ++n) {
    int e = label[n];
    e = (e < 0) ? 0 : ((e >= NEXP) ? (NEXP - 1) : e);
    perm[cur[e]++] = n;
  }
}

// ---------------- kernel 3: grouped QKV projection (WMMA) -----------------
// grid: (tokenTile<=256, outTile=16, expert=9), block = 32 (one wave)
__global__ void __launch_bounds__(32)
qkv_kernel(const _Float16* __restrict__ xh,
           const _Float16* __restrict__ wqT,
           const _Float16* __restrict__ wkT,
           const _Float16* __restrict__ wvT,
           const float* __restrict__ bq,
           const float* __restrict__ bk,
           const float* __restrict__ bv,
           const int* __restrict__ perm,
           const int* __restrict__ offs,
           _Float16* __restrict__ qs,
           _Float16* __restrict__ ks,
           _Float16* __restrict__ vTs) {
  int e  = blockIdx.z;
  int gs = offs[e], ge = offs[e + 1];
  int r0 = gs + (blockIdx.x << 4);
  if (r0 >= ge) return;
  int outBase = blockIdx.y << 4;
  int lane = threadIdx.x, lm = lane & 15, half = lane >> 4;

  int srow = r0 + lm; if (srow >= ge) srow = ge - 1;
  const _Float16* arow = xh + (size_t)perm[srow] * DMODEL;
  size_t wbase = (size_t)e * DMODEL * DMODEL + (size_t)(outBase + lm) * DMODEL;
  const _Float16* bqr = wqT + wbase;
  const _Float16* bkr = wkT + wbase;
  const _Float16* bvr = wvT + wbase;

  v8f cq = {}, ck = {}, cv = {};
  for (int k0 = 0; k0 < DMODEL; k0 += 32) {
    FragH a, fq, fk, fv;
#pragma unroll
    for (int j = 0; j < 8; ++j) {
      a.u[j]  = *(const unsigned int*)(arow + k0 + kA(j, half));
      fq.u[j] = *(const unsigned int*)(bqr  + k0 + kB(j, half));
      fk.u[j] = *(const unsigned int*)(bkr  + k0 + kB(j, half));
      fv.u[j] = *(const unsigned int*)(bvr  + k0 + kB(j, half));
    }
    cq = WMMA_F16(a.v, fq.v, cq);
    ck = WMMA_F16(a.v, fk.v, ck);
    cv = WMMA_F16(a.v, fv.v, cv);
  }
  // C layout: VGPR jc, row M = jc + half*8, col N = lm
#pragma unroll
  for (int jc = 0; jc < 8; ++jc) {
    int row = r0 + jc + (half << 3);
    if (row < ge) {
      int col = outBase + lm;
      float qv = cq[jc] + bq[e * DMODEL + col];
      float kv = ck[jc] + bk[e * DMODEL + col];
      float vv = cv[jc] + bv[e * DMODEL + col];
      qs[(size_t)row * DMODEL + col] = (_Float16)qv;
      ks[(size_t)row * DMODEL + col] = (_Float16)kv;
      vTs[(size_t)col * N_TOK + row] = (_Float16)vv;   // V transposed [D][N]
    }
  }
}

// ---------------- kernel 4: within-group flash attention (WMMA) -----------
// grid: (qTile<=256, head=4, expert=9), block = 32 (one wave)
__global__ void __launch_bounds__(32)
attn_kernel(const _Float16* __restrict__ qs,
            const _Float16* __restrict__ ks,
            const _Float16* __restrict__ vTs,
            const int* __restrict__ offs,
            _Float16* __restrict__ ctxs) {
  __shared__ _Float16 pbuf[16 * 34];   // 16 q-rows x 32 keys (+pad)
  int e  = blockIdx.z;
  int h  = blockIdx.y;
  int gs = offs[e], ge = offs[e + 1];
  int r0 = gs + (blockIdx.x << 4);
  if (r0 >= ge) return;
  int lane = threadIdx.x, lm = lane & 15, half = lane >> 4;

  int qrow = r0 + lm; if (qrow >= ge) qrow = ge - 1;
  const _Float16* qrp = qs + (size_t)qrow * DMODEL + h * DPH;

  v8f acc[4]; acc[0] = (v8f){}; acc[1] = (v8f){}; acc[2] = (v8f){}; acc[3] = (v8f){};
  float mrow[8], lrow[8];
#pragma unroll
  for (int jc = 0; jc < 8; ++jc) { mrow[jc] = -1e30f; lrow[jc] = 0.0f; }

  for (int kt = gs; kt < ge; kt += 32) {
    int key0 = kt + lm, key1 = key0 + 16;
    int kc0 = min(key0, ge - 1), kc1 = min(key1, ge - 1);
    const _Float16* kr0 = ks + (size_t)kc0 * DMODEL + h * DPH;
    const _Float16* kr1 = ks + (size_t)kc1 * DMODEL + h * DPH;

    v8f S0 = {}, S1 = {};
#pragma unroll
    for (int c0 = 0; c0 < DPH; c0 += 32) {
      FragH aq, b0, b1;
#pragma unroll
      for (int j = 0; j < 8; ++j) {
        aq.u[j] = *(const unsigned int*)(qrp + c0 + kA(j, half));
        b0.u[j] = *(const unsigned int*)(kr0 + c0 + kB(j, half));
        b1.u[j] = *(const unsigned int*)(kr1 + c0 + kB(j, half));
      }
      S0 = WMMA_F16(aq.v, b0.v, S0);
      S1 = WMMA_F16(aq.v, b1.v, S1);
    }
    if (kt + 32 < ge)
      __builtin_prefetch(ks + (size_t)min(kt + 32 + lm, ge - 1) * DMODEL + h * DPH, 0, 1);

    bool v0 = key0 < ge, v1 = key1 < ge;
    float p0[8], p1[8];
#pragma unroll
    for (int jc = 0; jc < 8; ++jc) {
      float s0 = v0 ? S0[jc] * ATT_SCALE : -1e30f;
      float s1 = v1 ? S1[jc] * ATT_SCALE : -1e30f;
      float t  = redmax16(fmaxf(s0, s1));
      float mn = fmaxf(mrow[jc], t);
      float alpha = __expf(mrow[jc] - mn);
      mrow[jc] = mn;
      float e0 = __expf(s0 - mn), e1 = __expf(s1 - mn);
      p0[jc] = e0; p1[jc] = e1;
      lrow[jc] = lrow[jc] * alpha + redsum16(e0 + e1);
      acc[0][jc] *= alpha; acc[1][jc] *= alpha;
      acc[2][jc] *= alpha; acc[3][jc] *= alpha;
    }

    __syncthreads();   // WAR vs previous iteration's pbuf reads
#pragma unroll
    for (int jc = 0; jc < 8; ++jc) {
      int M = jc + (half << 3);
      pbuf[M * 34 + lm]      = (_Float16)p0[jc];
      pbuf[M * 34 + 16 + lm] = (_Float16)p1[jc];
    }
    __syncthreads();   // RAW: P visible to whole wave

    FragH pa;
#pragma unroll
    for (int j = 0; j < 8; ++j)
      pa.u[j] = *(const unsigned int*)((const _Float16*)pbuf + lm * 34 + kA(j, half));

#pragma unroll
    for (int t = 0; t < 4; ++t) {
      const _Float16* vr = vTs + (size_t)(h * DPH + t * 16 + lm) * N_TOK;
      FragH bv_;
#pragma unroll
      for (int j = 0; j < 8; ++j) {
        int key = kt + kB(j, half);
        key = min(key, N_TOK - 2);     // padded allocation; invalid keys have P=0
        bv_.u[j] = *(const unsigned int*)(vr + key);
      }
      acc[t] = WMMA_F16(pa.v, bv_.v, acc[t]);
    }
  }

#pragma unroll
  for (int jc = 0; jc < 8; ++jc) {
    int row = r0 + jc + (half << 3);
    if (row < ge) {
      float inv = 1.0f / lrow[jc];
#pragma unroll
      for (int t = 0; t < 4; ++t)
        ctxs[(size_t)row * DMODEL + h * DPH + t * 16 + lm] = (_Float16)(acc[t][jc] * inv);
    }
  }
}

// ---------------- kernel 5: output projection + residual (WMMA) -----------
// grid: (256 token tiles, 16 out tiles), block = 32
__global__ void __launch_bounds__(32)
proj_kernel(const _Float16* __restrict__ ctxs,
            const _Float16* __restrict__ wfT,
            const float* __restrict__ bf,
            const float* __restrict__ x,
            const int* __restrict__ perm,
            float* __restrict__ hbuf) {
  int r0 = blockIdx.x << 4;
  int outBase = blockIdx.y << 4;
  int lane = threadIdx.x, lm = lane & 15, half = lane >> 4;
  const _Float16* arow = ctxs + (size_t)(r0 + lm) * DMODEL;
  const _Float16* brow = wfT + (size_t)(outBase + lm) * DMODEL;

  v8f c = {};
  for (int k0 = 0; k0 < DMODEL; k0 += 32) {
    FragH a, b;
#pragma unroll
    for (int j = 0; j < 8; ++j) {
      a.u[j] = *(const unsigned int*)(arow + k0 + kA(j, half));
      b.u[j] = *(const unsigned int*)(brow + k0 + kB(j, half));
    }
    c = WMMA_F16(a.v, b.v, c);
  }
#pragma unroll
  for (int jc = 0; jc < 8; ++jc) {
    int srow = r0 + jc + (half << 3);
    int orig = perm[srow];
    int col  = outBase + lm;
    float hval = c[jc] + bf[col] + x[(size_t)orig * DMODEL + col];
    hbuf[(size_t)orig * DMODEL + col] = hval;
  }
}

// ---------------- kernel 6: LayerNorm (one wave per token) ----------------
__global__ void ln_kernel(const float* __restrict__ hbuf,
                          const float* __restrict__ gamma,
                          const float* __restrict__ beta,
                          float* __restrict__ out) {
  int wid  = (blockIdx.x * blockDim.x + threadIdx.x) >> 5;
  int lane = threadIdx.x & 31;
  if (wid >= N_TOK) return;
  const float* hr = hbuf + (size_t)wid * DMODEL;
  float v[8], s = 0.0f;
#pragma unroll
  for (int i = 0; i < 8; ++i) { v[i] = hr[lane + i * 32]; s += v[i]; }
  s = redsum32(s);
  float mean = s * (1.0f / DMODEL);
  float vs = 0.0f;
#pragma unroll
  for (int i = 0; i < 8; ++i) { float d = v[i] - mean; vs += d * d; }
  vs = redsum32(vs);
  float rstd = rsqrtf(vs * (1.0f / DMODEL) + LN_EPS);
  float* orow = out + (size_t)wid * DMODEL;
#pragma unroll
  for (int i = 0; i < 8; ++i) {
    int cidx = lane + i * 32;
    orow[cidx] = gamma[cidx] * (v[i] - mean) * rstd + beta[cidx];
  }
}

// ---------------- host launcher -------------------------------------------
extern "C" void kernel_launch(void* const* d_in, const int* in_sizes, int n_in,
                              void* d_out, int out_size, void* d_ws, size_t ws_size,
                              hipStream_t stream) {
  const float* x     = (const float*)d_in[0];
  const int*   label = (const int*)d_in[1];
  const float* Wq    = (const float*)d_in[2];
  const float* bq    = (const float*)d_in[3];
  const float* Wk    = (const float*)d_in[4];
  const float* bk    = (const float*)d_in[5];
  const float* Wv    = (const float*)d_in[6];
  const float* bv    = (const float*)d_in[7];
  const float* Wf    = (const float*)d_in[8];
  const float* bf    = (const float*)d_in[9];
  const float* gamma = (const float*)d_in[10];
  const float* beta  = (const float*)d_in[11];
  float* out = (float*)d_out;

  // workspace carve-up (bytes)
  char* base = (char*)d_ws;
  size_t o = 0;
  _Float16* xh  = (_Float16*)(base + o); o += (size_t)N_TOK * DMODEL * 2;          // 2 MB
  _Float16* wqT = (_Float16*)(base + o); o += (size_t)NEXP * DMODEL * DMODEL * 2;  // 1.125 MB
  _Float16* wkT = (_Float16*)(base + o); o += (size_t)NEXP * DMODEL * DMODEL * 2;
  _Float16* wvT = (_Float16*)(base + o); o += (size_t)NEXP * DMODEL * DMODEL * 2;
  _Float16* wfT = (_Float16*)(base + o); o += (size_t)DMODEL * DMODEL * 2;
  _Float16* qs  = (_Float16*)(base + o); o += (size_t)N_TOK * DMODEL * 2;
  _Float16* ks  = (_Float16*)(base + o); o += (size_t)N_TOK * DMODEL * 2;
  _Float16* vTs = (_Float16*)(base + o); o += (size_t)DMODEL * N_TOK * 2 + 256;    // + overread pad
  _Float16* ctxs= (_Float16*)(base + o); o += (size_t)N_TOK * DMODEL * 2;
  float*    hbuf= (float*)   (base + o); o += (size_t)N_TOK * DMODEL * 4;          // 4 MB
  int*      perm= (int*)     (base + o); o += (size_t)N_TOK * 4;
  int*      offs= (int*)     (base + o); o += 64;
  // total ~17.6 MB

  prep_kernel<<<(N_TOK * DMODEL + 255) / 256, 256, 0, stream>>>(
      x, Wq, Wk, Wv, Wf, xh, wqT, wkT, wvT, wfT);

  sort_kernel<<<1, 32, 0, stream>>>(label, perm, offs);

  dim3 gq(N_TOK / 16, DMODEL / 16, NEXP);
  qkv_kernel<<<gq, 32, 0, stream>>>(xh, wqT, wkT, wvT, bq, bk, bv, perm, offs, qs, ks, vTs);

  dim3 ga(N_TOK / 16, NHEAD, NEXP);
  attn_kernel<<<ga, 32, 0, stream>>>(qs, ks, vTs, offs, ctxs);

  dim3 gp(N_TOK / 16, DMODEL / 16);
  proj_kernel<<<gp, 32, 0, stream>>>(ctxs, wfT, bf, x, perm, hbuf);

  ln_kernel<<<(N_TOK * 32 + 255) / 256, 256, 0, stream>>>(hbuf, gamma, beta, out);
}
